// BidirectionalLayer_77592879170085
// MI455X (gfx1250) — matrix operand order repaired and checked by reading, not compiled, not run on hardware
//
#include <hip/hip_runtime.h>
#include <hip/hip_bf16.h>

// ---------------------------------------------------------------------------
// Types for CDNA5 WMMA (wave32): D(16x16 f32) = A(16x32 f16) x B(32x16 f16) + C
// ---------------------------------------------------------------------------
typedef __attribute__((ext_vector_type(16))) _Float16 v16h;
typedef __attribute__((ext_vector_type(8)))  float    v8f;
typedef __attribute__((ext_vector_type(4)))  float    f32x4;

union FragU {
    v16h  h;
    f32x4 f[2];
};

__device__ __forceinline__ float sigf(float x)      { return 1.f / (1.f + __expf(-x)); }
__device__ __forceinline__ float softplusf(float x) { return (x > 20.f) ? x : log1pf(__expf(x)); }

// ---------------------------------------------------------------------------
// Weight packing: f32 row-major [Ka+Kb, Nsrc] -> f16 fragment-major.
// B-fragment layout (16-bit B 32x16): lane l (c=l&15, hi=l>>4), half h:
//   K = kt*32 + hi*16 + h ; N = nt*16 + c
// dst index = ((kt*Ntiles + nt)*512 + l*16 + h)
// ---------------------------------------------------------------------------
__global__ __launch_bounds__(256) void pack_weight(
    const float* __restrict__ srcA, int Ka,
    const float* __restrict__ srcB, int Kb,
    int Nsrc, int Ntiles, _Float16* __restrict__ dst, int total)
{
    int gid = blockIdx.x * 256 + threadIdx.x;
    if (gid >= total) return;
    int h    = gid & 15;
    int l    = (gid >> 4) & 31;
    int frag = gid >> 9;
    int nt   = frag % Ntiles;
    int kt   = frag / Ntiles;
    int c = l & 15, hi = l >> 4;
    int k = kt * 32 + hi * 16 + h;
    int n = nt * 16 + c;
    float v = 0.f;
    if (n < Nsrc)
        v = (k < Ka) ? srcA[(size_t)k * Nsrc + n] : srcB[(size_t)(k - Ka) * Nsrc + n];
    dst[gid] = (_Float16)v;
}

// f32 rows -> f16 rows (for staging x_t into the activation concat buffers)
__global__ __launch_bounds__(256) void cast_rows_f16(
    const float* __restrict__ src, long long srcStride,
    _Float16* __restrict__ dst, int dld, int cols)
{
    int gid = blockIdx.x * 256 + threadIdx.x;
    int m = gid / cols, c = gid % cols;
    if (m >= 64) return;
    dst[(size_t)m * dld + c] = (_Float16)src[(size_t)m * srcStride + c];
}

// ---------------------------------------------------------------------------
// GEMM: C(64 x Ntiles*16, f32) = [A0 | A1](64 x K0+K1, f16 row-major) @ Bpacked
// One block = 8 waves; wave w owns N-tile (blockIdx.x*8+w) across 4 M-tiles.
// A fragment (16-bit A 16x32): lane l (r=l&15, hi=l>>4):
//   halves 0..7  : K = kbase + hi*16 + h       (contiguous 16B)
//   halves 8..15 : K = kbase + hi*16 + h       (contiguous 16B)
// ---------------------------------------------------------------------------
__global__ __launch_bounds__(256) void gemm_f16_wmma(
    const _Float16* __restrict__ A0, int ld0, int K0,
    const _Float16* __restrict__ A1, int ld1, int K1,
    const _Float16* __restrict__ Bp,
    float* __restrict__ C, int ldc, int Ntiles)
{
    int wave = threadIdx.x >> 5;
    int lane = threadIdx.x & 31;
    int nt = blockIdx.x * 8 + wave;
    if (nt >= Ntiles) return;               // wave-uniform: EXEC stays all-1s for WMMA
    int r = lane & 15, hi = lane >> 4;
    int Ktiles = (K0 + K1) >> 5;

    v8f acc[4] = {};
    for (int kt = 0; kt < Ktiles; ++kt) {
        int kbase = kt << 5;
        const _Float16* As; int ldA, kl;
        if (kbase < K0) { As = A0; ldA = ld0; kl = kbase; }
        else            { As = A1; ldA = ld1; kl = kbase - K0; }

        FragU b;
        const f32x4* bp = (const f32x4*)(Bp + (((size_t)(kt * Ntiles + nt)) << 9) + (lane << 4));
        b.f[0] = bp[0];
        b.f[1] = bp[1];

        int off0 = kl + hi * 16;
#pragma unroll
        for (int mt = 0; mt < 4; ++mt) {
            FragU a;
            const _Float16* ap = As + (size_t)(mt * 16 + r) * ldA + off0;
            a.f[0] = *(const f32x4*)ap;
            a.f[1] = *(const f32x4*)(ap + 8);
            acc[mt] = __builtin_amdgcn_wmma_f32_16x16x32_f16(
                false, a.h, false, b.h, (short)0, acc[mt], false, false);
        }
    }
    // C/D layout: VGPR v, lanes 0-15 -> M=v, lanes 16-31 -> M=v+8 ; N = lane&15
    int col = nt * 16 + r;
    int rb  = hi * 8;
#pragma unroll
    for (int mt = 0; mt < 4; ++mt)
#pragma unroll
        for (int v = 0; v < 8; ++v)
            C[(size_t)(mt * 16 + rb + v) * ldc + col] = acc[mt][v];
}

// ---------------------------------------------------------------------------
// LSTM pointwise: z(64x2048 f32) [i|f|g|o], bias, c state (in-place),
// h written as f16 to two destinations (concat slots of downstream GEMM A's).
// ---------------------------------------------------------------------------
__global__ __launch_bounds__(256) void lstm_pointwise(
    const float* __restrict__ z, const float* __restrict__ bias,
    float* __restrict__ c,
    _Float16* __restrict__ d0, int ld0,
    _Float16* __restrict__ d1, int ld1)
{
    int idx = blockIdx.x * 256 + threadIdx.x;
    if (idx >= 64 * 512) return;
    int m = idx >> 9, u = idx & 511;
    const float* zr = z + (size_t)m * 2048;
    float i = sigf(zr[u]          + bias[u]);
    float f = sigf(zr[512 + u]    + bias[512 + u]);
    float g = tanhf(zr[1024 + u]  + bias[1024 + u]);
    float o = sigf(zr[1536 + u]   + bias[1536 + u]);
    float cc = f * c[idx] + i * g;
    c[idx] = cc;
    _Float16 hh = (_Float16)(o * tanhf(cc));
    d0[(size_t)m * ld0 + u] = hh;
    d1[(size_t)m * ld1 + u] = hh;
}

// ---------------------------------------------------------------------------
// DNC memory step: one block per batch element, 256 threads. M staged in LDS.
// N=128, W=64, R=4. xi buffer is 64x480 (cols 471..479 are zero pads).
// ---------------------------------------------------------------------------
__global__ __launch_bounds__(256) void dnc_step(
    const float* __restrict__ xiG, const float* __restrict__ bif,
    float* __restrict__ Mg, float* __restrict__ Lg,
    float* __restrict__ ug, float* __restrict__ wwg, float* __restrict__ pg,
    float* __restrict__ wrg,
    float* __restrict__ outp, _Float16* __restrict__ readA, int ldA)
{
    __shared__ float sM[128 * 64];
    __shared__ float sxi[480];
    __shared__ float su[128], sa[128], sww[128], sp[128];
    __shared__ float scw[128];
    __shared__ float scr[4][128];
    __shared__ float swr[4][128], swrn[4][128];
    __shared__ float sfwd[4][128], sbwd[4][128];
    __shared__ float sred[4];

    const int b = blockIdx.x, tid = threadIdx.x;
    const float* xi = xiG + (size_t)b * 480;
    float* Mp = Mg + (size_t)b * 8192;
    float* Lp = Lg + (size_t)b * 16384;

    for (int i = tid; i < 480; i += 256) sxi[i] = (i < 471) ? (xi[i] + bif[i]) : 0.f;
    for (int i = tid; i < 8192; i += 256) sM[i] = Mp[i];
    for (int i = tid; i < 512; i += 256) swr[i >> 7][i & 127] = wrg[(size_t)b * 512 + i];
    if (tid < 128) { su[tid] = ug[b*128+tid]; sww[tid] = wwg[b*128+tid]; sp[tid] = pg[b*128+tid]; }
    __syncthreads();

    // psi & usage update (uses OLD wr, OLD ww)
    if (tid < 128) {
        int n = tid;
        float psi = 1.f;
#pragma unroll
        for (int r = 0; r < 4; ++r) {
            float fr = sigf(sxi[453 + r]);
            psi *= (1.f - fr * swr[r][n]);
        }
        float uu = su[n], w = sww[n];
        su[n] = (uu + w - uu * w) * psi;
    }
    __syncthreads();

    // allocation weights (sort-free; stable-ascending tie-break == jnp.argsort)
    if (tid < 128) {
        int n = tid;
        float un = su[n], prod = 1.f;
        for (int k = 0; k < 128; ++k) {
            float uk = su[k];
            bool before = (uk < un) || (uk == un && k < n);
            prod *= before ? uk : 1.f;
        }
        sa[n] = (1.f - un) * prod;
    }
    // write-content logits on OLD M
    if (tid < 128) {
        int n = tid;
        float dot = 0.f, nrm = 0.f, knrm = 0.f;
        for (int w = 0; w < 64; ++w) {
            float m = sM[n * 64 + w], kw = sxi[260 + w];
            dot += m * kw; nrm += m * m; knrm += kw * kw;
        }
        float beta_w = 1.f + softplusf(sxi[324]);
        scw[n] = beta_w * dot / (sqrtf(nrm) * sqrtf(knrm) + 1e-6f);
    }
    __syncthreads();
    if (tid == 0) { // softmax over N (serial; trivial cost)
        float mx = scw[0]; for (int i = 1; i < 128; ++i) mx = fmaxf(mx, scw[i]);
        float s = 0.f; for (int i = 0; i < 128; ++i) { float e = __expf(scw[i] - mx); scw[i] = e; s += e; }
        float inv = 1.f / s; for (int i = 0; i < 128; ++i) scw[i] *= inv;
    }
    __syncthreads();
    if (tid < 128) { // write weights
        float g_a = sigf(sxi[457]), g_w = sigf(sxi[458]);
        sww[tid] = g_w * (g_a * sa[tid] + (1.f - g_a) * scw[tid]);
    }
    __syncthreads();
    // M update (erase/write)
    for (int i = tid; i < 8192; i += 256) {
        int n = i >> 6, w = i & 63;
        float e = sigf(sxi[325 + w]), v = sxi[389 + w];
        sM[i] = sM[i] * (1.f - sww[n] * e) + sww[n] * v;
    }
    // L update (uses new ww, old p); zero diagonal
    for (int i = tid; i < 16384; i += 256) {
        int ii = i >> 7, jj = i & 127;
        float lv = (ii == jj) ? 0.f
                 : (1.f - sww[ii] - sww[jj]) * Lp[i] + sww[ii] * sp[jj];
        Lp[i] = lv;
    }
    __threadfence_block();
    __syncthreads();
    // precedence update
    if (tid == 0) { float s = 0.f; for (int i = 0; i < 128; ++i) s += sww[i]; sred[0] = s; }
    __syncthreads();
    if (tid < 128) sp[tid] = (1.f - sred[0]) * sp[tid] + sww[tid];
    // read-content logits on NEW M
    if (tid < 128) {
        int n = tid;
        float dot[4] = {0,0,0,0}, knrm[4] = {0,0,0,0}, nrm = 0.f;
        for (int w = 0; w < 64; ++w) {
            float m = sM[n * 64 + w];
            nrm += m * m;
#pragma unroll
            for (int r = 0; r < 4; ++r) { float kr = sxi[r * 64 + w]; dot[r] += m * kr; knrm[r] += kr * kr; }
        }
        float nr = sqrtf(nrm);
#pragma unroll
        for (int r = 0; r < 4; ++r) {
            float beta = 1.f + softplusf(sxi[256 + r]);
            scr[r][n] = beta * dot[r] / (nr * sqrtf(knrm[r]) + 1e-6f);
        }
    }
    __syncthreads();
    if (tid < 4) { // softmax per read head
        int r = tid;
        float mx = scr[r][0]; for (int i = 1; i < 128; ++i) mx = fmaxf(mx, scr[r][i]);
        float s = 0.f; for (int i = 0; i < 128; ++i) { float e = __expf(scr[r][i] - mx); scr[r][i] = e; s += e; }
        float inv = 1.f / s; for (int i = 0; i < 128; ++i) scr[r][i] *= inv;
    }
    __syncthreads();
    // forward/backward linkage reads (new L, OLD wr)
    for (int rr = 0; rr < 2; ++rr) {
        int r = rr * 2 + (tid >> 7);
        int n = tid & 127;
        float f = 0.f, bw = 0.f;
        for (int j = 0; j < 128; ++j) {
            float w = swr[r][j];
            f  += Lp[n * 128 + j] * w;
            bw += Lp[j * 128 + n] * w;
        }
        sfwd[r][n] = f; sbwd[r][n] = bw;
    }
    __syncthreads();
    // new read weights
    for (int i = tid; i < 512; i += 256) {
        int r = i >> 7, n = i & 127;
        float e0 = sxi[459 + r * 3], e1 = sxi[460 + r * 3], e2 = sxi[461 + r * 3];
        float mx = fmaxf(e0, fmaxf(e1, e2));
        float x0 = __expf(e0 - mx), x1 = __expf(e1 - mx), x2 = __expf(e2 - mx);
        float inv = 1.f / (x0 + x1 + x2);
        swrn[r][n] = (x0 * sbwd[r][n] + x1 * scr[r][n] + x2 * sfwd[r][n]) * inv;
    }
    __syncthreads();
    // read vectors -> output (f32) and next-step layer0 input slot (f16)
    {
        int r = tid >> 6, w = tid & 63;
        float acc = 0.f;
        for (int n = 0; n < 128; ++n) acc += swrn[r][n] * sM[n * 64 + w];
        outp[(size_t)b * 256 + r * 64 + w] = acc;
        readA[(size_t)b * ldA + r * 64 + w] = (_Float16)acc;
    }
    // state writeback
    for (int i = tid; i < 8192; i += 256) Mp[i] = sM[i];
    if (tid < 128) { ug[b*128+tid] = su[tid]; wwg[b*128+tid] = sww[tid]; pg[b*128+tid] = sp[tid]; }
    for (int i = tid; i < 512; i += 256) wrg[(size_t)b * 512 + i] = swrn[i >> 7][i & 127];
}

// ---------------------------------------------------------------------------
// Host orchestration
// ---------------------------------------------------------------------------
extern "C" void kernel_launch(void* const* d_in, const int* in_sizes, int n_in,
                              void* d_out, int out_size, void* d_ws, size_t ws_size,
                              hipStream_t stream)
{
    (void)in_sizes; (void)n_in; (void)out_size; (void)ws_size;
    const float* x      = (const float*)d_in[0];
    const float* Wx0_fw = (const float*)d_in[1];
    const float* Wh0_fw = (const float*)d_in[2];
    const float* b0_fw  = (const float*)d_in[3];
    const float* Wx1_fw = (const float*)d_in[4];
    const float* Wh1_fw = (const float*)d_in[5];
    const float* b1_fw  = (const float*)d_in[6];
    const float* Wx0_bw = (const float*)d_in[7];
    const float* Wh0_bw = (const float*)d_in[8];
    const float* b0_bw  = (const float*)d_in[9];
    const float* Wx1_bw = (const float*)d_in[10];
    const float* Wh1_bw = (const float*)d_in[11];
    const float* b1_bw  = (const float*)d_in[12];
    const float* W_if   = (const float*)d_in[13];
    const float* b_if   = (const float*)d_in[14];
    float* out = (float*)d_out;

    char* ws = (char*)d_ws;
    size_t off = 0;
    auto carve = [&](size_t bytes) -> char* {
        char* p = ws + off;
        off += (bytes + 255) & ~(size_t)255;
        return p;
    };

    // ---- zero-initialized region (states + activation concat buffers) ----
    float*    M    = (float*)carve(64ull * 128 * 64 * 4);     // memory
    float*    L    = (float*)carve(64ull * 128 * 128 * 4);    // linkage
    float*    u    = (float*)carve(64ull * 128 * 4);
    float*    wwS  = (float*)carve(64ull * 128 * 4);
    float*    p    = (float*)carve(64ull * 128 * 4);
    float*    wr   = (float*)carve(64ull * 4 * 128 * 4);
    float*    c0f  = (float*)carve(64ull * 512 * 4);
    float*    c1f  = (float*)carve(64ull * 512 * 4);
    float*    c0b  = (float*)carve(64ull * 512 * 4);
    float*    c1b  = (float*)carve(64ull * 512 * 4);
    _Float16* actA = (_Float16*)carve(64ull * 1024 * 2);      // [x|read|h0] fw L0
    _Float16* actB = (_Float16*)carve(64ull * 1024 * 2);      // [h0|h1]     fw L1
    _Float16* actC = (_Float16*)carve(64ull * 768 * 2);       // [x|h0]      bw L0
    _Float16* actD = (_Float16*)carve(64ull * 1024 * 2);      // [h0|h1]     bw L1
    size_t zeroBytes = off;

    // ---- scratch (always written before read) ----
    float*    z0    = (float*)carve(64ull * 2048 * 4);
    float*    z1    = (float*)carve(64ull * 2048 * 4);
    float*    xi    = (float*)carve(64ull * 480 * 4);
    _Float16* muin  = (_Float16*)carve(64ull * 512 * 2);      // h1 half of mu input
    _Float16* outbw = (_Float16*)carve(512ull * 64 * 512 * 2);
    _Float16* Wp0f  = (_Float16*)carve(32ull * 128 * 512 * 2);
    _Float16* Wp1f  = (_Float16*)carve(32ull * 128 * 512 * 2);
    _Float16* Wp0b  = (_Float16*)carve(24ull * 128 * 512 * 2);
    _Float16* Wp1b  = (_Float16*)carve(32ull * 128 * 512 * 2);
    _Float16* Wpif  = (_Float16*)carve(32ull * 30 * 512 * 2);

    hipMemsetAsync(ws, 0, zeroBytes, stream);

    // ---- pack weights (fragment-major f16), done per call (deterministic) ----
    {
        int tot = 32 * 128 * 512;
        pack_weight<<<tot / 256, 256, 0, stream>>>(Wx0_fw, 512, Wh0_fw, 512, 2048, 128, Wp0f, tot);
        pack_weight<<<tot / 256, 256, 0, stream>>>(Wx1_fw, 512, Wh1_fw, 512, 2048, 128, Wp1f, tot);
        pack_weight<<<tot / 256, 256, 0, stream>>>(Wx1_bw, 512, Wh1_bw, 512, 2048, 128, Wp1b, tot);
        int totb = 24 * 128 * 512;
        pack_weight<<<totb / 256, 256, 0, stream>>>(Wx0_bw, 256, Wh0_bw, 512, 2048, 128, Wp0b, totb);
        int toti = 32 * 30 * 512;
        pack_weight<<<toti / 256, 256, 0, stream>>>(W_if, 1024, W_if, 0, 471, 30, Wpif, toti);
    }

    const long long xStride = 512ll * 256;   // (B,T,F): row stride over b

    // ---- backward LSTM scan (processes x[T-1-t], emits out_bw[t]) ----
    for (int t = 0; t < 512; ++t) {
        cast_rows_f16<<<64, 256, 0, stream>>>(x + (size_t)(511 - t) * 256, xStride, actC, 768, 256);
        gemm_f16_wmma<<<16, 256, 0, stream>>>(actC, 768, 768, actC, 0, 0, Wp0b, z0, 2048, 128);
        lstm_pointwise<<<128, 256, 0, stream>>>(z0, b0_bw, c0b, actD, 1024, actC + 256, 768);
        gemm_f16_wmma<<<16, 256, 0, stream>>>(actD, 1024, 1024, actD, 0, 0, Wp1b, z1, 2048, 128);
        lstm_pointwise<<<128, 256, 0, stream>>>(z1, b1_bw, c1b,
                                                actD + 512, 1024,
                                                outbw + (size_t)t * 64 * 512, 512);
    }

    // ---- forward scan: LSTM x2 -> interface GEMM -> DNC memory step ----
    for (int t = 0; t < 512; ++t) {
        cast_rows_f16<<<64, 256, 0, stream>>>(x + (size_t)t * 256, xStride, actA, 1024, 256);
        gemm_f16_wmma<<<16, 256, 0, stream>>>(actA, 1024, 1024, actA, 0, 0, Wp0f, z0, 2048, 128);
        lstm_pointwise<<<128, 256, 0, stream>>>(z0, b0_fw, c0f, actB, 1024, actA + 512, 1024);
        gemm_f16_wmma<<<16, 256, 0, stream>>>(actB, 1024, 1024, actB, 0, 0, Wp1f, z1, 2048, 128);
        lstm_pointwise<<<128, 256, 0, stream>>>(z1, b1_fw, c1f, actB + 512, 1024, muin, 512);
        // xi = [h1 | out_bw[t]] @ W_if   (two A segments, no copy)
        gemm_f16_wmma<<<4, 256, 0, stream>>>(muin, 512, 512,
                                             outbw + (size_t)t * 64 * 512, 512, 512,
                                             Wpif, xi, 480, 30);
        dnc_step<<<64, 256, 0, stream>>>(xi, b_if, M, L, u, wwS, p, wr,
                                         out + (size_t)t * 64 * 256,
                                         actA + 256, 1024);
    }
}
